// StateSpaceLayer_37014028157374
// MI455X (gfx1250) — matrix operand order repaired and checked by reading, not compiled, not run on hardware
//
#include <hip/hip_runtime.h>

// ---------------------------------------------------------------------------
// StateSpaceLayer for MI455X (gfx1250, wave32, WMMA bf16 -> f32 accumulate)
//   B=128, T=256, D=768, S=128
//   k_cvt : f32 -> bf16 weight conversion (A, B_w, C_w, D_w)
//   k_bx  : Bx = x @ B_w^T                          (WMMA)
//   k_scan: state_t = clip(state @ A^T + Bx_t)      (WMMA, serial in T)
//   k_out : out = LN(gate0*(states@C^T + x@D^T) + gate1*x)  (WMMA + fused)
// ---------------------------------------------------------------------------

#define DI 768
#define SD 128
#define BB 128
#define TT 256
#define M_ALL (BB * TT)

typedef __attribute__((ext_vector_type(16))) __bf16 bf16x16;
typedef __attribute__((ext_vector_type(8)))  float  f32x8;

union Frag { bf16x16 v; uint4 q[2]; };

__device__ __forceinline__ unsigned short f2bf(float x) {
    unsigned u = __float_as_uint(x);
    u += 0x7FFFu + ((u >> 16) & 1u);        // round-to-nearest-even
    return (unsigned short)(u >> 16);
}
__device__ __forceinline__ float bf2f(unsigned short h) {
    return __uint_as_float(((unsigned)h) << 16);
}
__device__ __forceinline__ f32x8 zero8() {
    f32x8 z;
#pragma unroll
    for (int i = 0; i < 8; ++i) z[i] = 0.f;
    return z;
}
__device__ __forceinline__ f32x8 wmma_bf16(const Frag& a, const Frag& b, f32x8 c) {
    return __builtin_amdgcn_wmma_f32_16x16x32_bf16(
        /*neg_a=*/false, a.v, /*neg_b=*/false, b.v,
        /*c_mod=*/(short)0, c, /*reuse_a=*/false, /*reuse_b=*/false);
}

// A-operand (16xK, 16-bit, ISA 7.12.2): lane holds row = lane%16;
// VGPR0-3 = K[k0 + (lane/16)*8 .. +7], VGPR4-7 = K[k0+16 + (lane/16)*8 .. +7]
__device__ __forceinline__ Frag load_a(const unsigned short* base, int stride,
                                       int k0, int lane) {
    Frag f;
    const unsigned short* p =
        base + (lane & 15) * stride + k0 + ((lane >> 4) << 3);
    f.q[0] = *(const uint4*)(p);
    f.q[1] = *(const uint4*)(p + 16);
    return f;
}

// B-operand (Kx16 tile of W^T with W stored [N][K] row-major):
// lane holds col N = n0 + lane%16, K = k0 + (lane/16)*8 .. +7, k0+16 + (lane/16)*8 .. +7
__device__ __forceinline__ Frag load_b(const unsigned short* W, int ldk,
                                       int n0, int k0, int lane) {
    Frag f;
    const unsigned short* p =
        W + (size_t)(n0 + (lane & 15)) * ldk + k0 + ((lane >> 4) << 3);
    f.q[0] = *(const uint4*)(p);
    f.q[1] = *(const uint4*)(p + 16);
    return f;
}

// pack 4 consecutive f32 -> 4 bf16 (uint2)
__device__ __forceinline__ uint2 pack4(float4 v) {
    uint2 r;
    r.x = ((unsigned)f2bf(v.y) << 16) | f2bf(v.x);
    r.y = ((unsigned)f2bf(v.w) << 16) | f2bf(v.z);
    return r;
}

// ---------------------------------------------------------------------------
__global__ void k_cvt(const float* __restrict__ s,
                      unsigned short* __restrict__ d, int n4) {
    int i = blockIdx.x * blockDim.x + threadIdx.x;
    if (i < n4) {
        float4 v = *(const float4*)(s + (size_t)i * 4);
        *(uint2*)(d + (size_t)i * 4) = pack4(v);
    }
}

// ---------------------------------------------------------------------------
// Bx = x @ B_w^T : M=32768, N=128, K=768. 16-row tile per block, 8 waves,
// one 16-col N-tile per wave.
__global__ __launch_bounds__(256) void k_bx(const float* __restrict__ x,
                                            const unsigned short* __restrict__ Bw,
                                            float* __restrict__ Bx) {
    __shared__ unsigned short xb[16][776];   // padded: breaks LDS bank conflicts
    const int tid  = threadIdx.x;
    const int lane = tid & 31;
    const int wv   = tid >> 5;               // 0..7
    const int m0   = blockIdx.x << 4;

    for (int i = tid; i < 16 * DI / 4; i += 256) {   // float4 -> 4xbf16
        int idx = i << 2;
        int r = idx / DI, c = idx - r * DI;
        float4 v = *(const float4*)(x + (size_t)(m0 + r) * DI + c);
        *(uint2*)&xb[r][c] = pack4(v);
    }
    __syncthreads();

    f32x8 acc = zero8();
    const int n0 = wv << 4;
#pragma unroll 4
    for (int kc = 0; kc < DI / 32; ++kc) {
        Frag a = load_a(&xb[0][0], 776, kc * 32, lane);
        Frag b = load_b(Bw, DI, n0, kc * 32, lane);
        acc = wmma_bf16(a, b, acc);
    }
    const int col = n0 + (lane & 15);
    const int mh  = (lane >> 4) << 3;
#pragma unroll
    for (int r = 0; r < 8; ++r)
        Bx[(size_t)(m0 + mh + r) * SD + col] = acc[r];
}

// ---------------------------------------------------------------------------
// Sequential scan. 8 blocks x 16 batch rows; A lives in LDS; state single-
// buffered via two barriers per step. States stored as bf16 for k_out.
__global__ __launch_bounds__(256) void k_scan(const unsigned short* __restrict__ Ab,
                                              const float* __restrict__ Bx,
                                              unsigned short* __restrict__ Sg) {
    __shared__ unsigned short Asm[128][136]; // A[n][k], padded stride
    __shared__ unsigned short st[16][136];   // state tile, bf16
    const int tid  = threadIdx.x;
    const int lane = tid & 31;
    const int wv   = tid >> 5;               // 0..7
    const int b0   = blockIdx.x << 4;        // batch row base

    for (int i = tid; i < 128 * 128 / 8; i += 256) {  // uint4 copies (8 bf16)
        int idx = i << 3;
        int r = idx >> 7, c = idx & 127;
        *(uint4*)&Asm[r][c] = *(const uint4*)(Ab + idx);
    }
    for (int i = tid; i < 16 * 136; i += 256)
        (&st[0][0])[i] = 0;                  // initial state = 0

    const int n0  = wv << 4;
    const int col = n0 + (lane & 15);
    const int mh  = (lane >> 4) << 3;

    for (int t = 0; t < TT; ++t) {
        __syncthreads();                     // state tile ready
        Frag a0 = load_a(&st[0][0], 136, 0,  lane);
        Frag a1 = load_a(&st[0][0], 136, 32, lane);
        Frag a2 = load_a(&st[0][0], 136, 64, lane);
        Frag a3 = load_a(&st[0][0], 136, 96, lane);

        f32x8 acc;                           // C init = Bx(t) fragment
#pragma unroll
        for (int r = 0; r < 8; ++r)
            acc[r] = Bx[((size_t)(b0 + mh + r) * TT + t) * SD + col];

        acc = wmma_bf16(a0, load_b(&Asm[0][0], 136, n0, 0,  lane), acc);
        acc = wmma_bf16(a1, load_b(&Asm[0][0], 136, n0, 32, lane), acc);
        acc = wmma_bf16(a2, load_b(&Asm[0][0], 136, n0, 64, lane), acc);
        acc = wmma_bf16(a3, load_b(&Asm[0][0], 136, n0, 96, lane), acc);

#pragma unroll
        for (int r = 0; r < 8; ++r)
            acc[r] = fminf(fmaxf(acc[r], -10.f), 10.f);

        __syncthreads();                     // everyone finished reading old state
#pragma unroll
        for (int r = 0; r < 8; ++r) {
            unsigned short h = f2bf(acc[r]);
            st[mh + r][col] = h;
            Sg[((size_t)(b0 + mh + r) * TT + t) * SD + col] = h;
        }
    }
}

// ---------------------------------------------------------------------------
// Fused: mm = states@C^T + x@D^T ; gate softmax ; mix ; LayerNorm ; store.
// 16-row tile, 16 waves (512 thr). Each wave: 3 N-tiles of 16 (48 total),
// K-outer loop shares each LDS A-fragment across the 3 WMMAs; then each
// wave owns one row for the gate + LayerNorm reductions.
__global__ __launch_bounds__(512) void k_out(const float* __restrict__ x,
                                             const unsigned short* __restrict__ Sg,
                                             const unsigned short* __restrict__ Cw,
                                             const unsigned short* __restrict__ Dw,
                                             const float* __restrict__ gw,
                                             const float* __restrict__ gb,
                                             const float* __restrict__ gamma,
                                             const float* __restrict__ beta,
                                             float* __restrict__ out) {
    __shared__ union { unsigned short xb[16][776]; float ot[16][776]; } sm;
    __shared__ unsigned short st[16][136];
    __shared__ float p0s[16], p1s[16];
    const int tid  = threadIdx.x;
    const int lane = tid & 31;
    const int wv   = tid >> 5;               // 0..15
    const int m0   = blockIdx.x << 4;

    for (int i = tid; i < 16 * DI / 4; i += 512) {   // float4 -> 4xbf16
        int idx = i << 2;
        int r = idx / DI, c = idx - r * DI;
        float4 v = *(const float4*)(x + (size_t)(m0 + r) * DI + c);
        *(uint2*)&sm.xb[r][c] = pack4(v);
    }
    for (int i = tid; i < 16 * SD / 8; i += 512) {   // uint4 copies (8 bf16)
        int idx = i << 3;
        int r = idx >> 7, c = idx & 127;
        *(uint4*)&st[r][c] = *(const uint4*)(Sg + (size_t)(m0 + r) * SD + c);
    }
    __syncthreads();

    // --- gate logits for row wv (bf16 x from LDS, f32 gate weights) ---
    {
        float g0 = 0.f, g1 = 0.f;
        for (int c = lane; c < DI; c += 32) {
            float xv = bf2f(sm.xb[wv][c]);
            g0 += xv * gw[c];
            g1 += xv * gw[DI + c];
        }
        for (int off = 16; off; off >>= 1) {
            g0 += __shfl_xor(g0, off);
            g1 += __shfl_xor(g1, off);
        }
        if (lane == 0) {
            g0 += gb[0]; g1 += gb[1];
            float mx = fmaxf(g0, g1);
            float e0 = __expf(g0 - mx), e1 = __expf(g1 - mx);
            float inv = 1.f / (e0 + e1);
            p0s[wv] = e0 * inv;
            p1s[wv] = e1 * inv;
        }
    }

    // --- GEMM: K-outer, A-fragment shared across 3 N-tiles per wave ---
    f32x8 acc[3];
    acc[0] = zero8(); acc[1] = zero8(); acc[2] = zero8();
    const int nb = wv << 4;                  // tile i -> n0 = nb + i*256
    for (int kc = 0; kc < DI / 32; ++kc) {   // x @ D_w^T (K = 768)
        Frag a = load_a(&sm.xb[0][0], 776, kc * 32, lane);
#pragma unroll
        for (int i = 0; i < 3; ++i)
            acc[i] = wmma_bf16(a, load_b(Dw, DI, nb + (i << 8), kc * 32, lane),
                               acc[i]);
    }
#pragma unroll
    for (int kc = 0; kc < SD / 32; ++kc) {   // states @ C_w^T (K = 128)
        Frag a = load_a(&st[0][0], 136, kc * 32, lane);
#pragma unroll
        for (int i = 0; i < 3; ++i)
            acc[i] = wmma_bf16(a, load_b(Cw, SD, nb + (i << 8), kc * 32, lane),
                               acc[i]);
    }
    __syncthreads();   // all xb reads done -> safe to overlay ot; p0s visible

    const int mh = (lane >> 4) << 3;
#pragma unroll
    for (int i = 0; i < 3; ++i) {
        const int cc = nb + (i << 8) + (lane & 15);
#pragma unroll
        for (int r = 0; r < 8; ++r)
            sm.ot[mh + r][cc] = p0s[mh + r] * acc[i][r];
    }
    __syncthreads();

    // --- residual mix + LayerNorm for row wv ---
    {
        const int gm = m0 + wv;
        const float p1 = p1s[wv];
        float vals[24];
        float s = 0.f, sq = 0.f;
#pragma unroll
        for (int j = 0; j < 24; ++j) {
            int c = lane + (j << 5);
            float v = sm.ot[wv][c] + p1 * x[(size_t)gm * DI + c];
            vals[j] = v; s += v; sq += v * v;
        }
        for (int off = 16; off; off >>= 1) {
            s  += __shfl_xor(s,  off);
            sq += __shfl_xor(sq, off);
        }
        const float mu   = s * (1.f / DI);
        const float var  = sq * (1.f / DI) - mu * mu;
        const float rstd = rsqrtf(var + 1e-5f);
#pragma unroll
        for (int j = 0; j < 24; ++j) {
            int c = lane + (j << 5);
            out[(size_t)gm * DI + c] = (vals[j] - mu) * rstd * gamma[c] + beta[c];
        }
    }
}

// ---------------------------------------------------------------------------
extern "C" void kernel_launch(void* const* d_in, const int* in_sizes, int n_in,
                              void* d_out, int out_size, void* d_ws, size_t ws_size,
                              hipStream_t stream) {
    const float* x     = (const float*)d_in[0];
    const float* A     = (const float*)d_in[1];
    const float* Bw    = (const float*)d_in[2];
    const float* Cw    = (const float*)d_in[3];
    const float* Dw    = (const float*)d_in[4];
    const float* gw    = (const float*)d_in[5];
    const float* gb    = (const float*)d_in[6];
    const float* gamma = (const float*)d_in[7];
    const float* beta  = (const float*)d_in[8];
    float* out = (float*)d_out;

    // workspace layout (256-byte aligned), ~26.8 MB total
    char* p = (char*)d_ws;
    auto take = [&](size_t bytes) {
        char* r = p;
        p += (bytes + 255) & ~(size_t)255;
        return r;
    };
    unsigned short* A_b  = (unsigned short*)take((size_t)SD * SD * 2);
    unsigned short* Bw_b = (unsigned short*)take((size_t)SD * DI * 2);
    unsigned short* Cw_b = (unsigned short*)take((size_t)DI * SD * 2);
    unsigned short* Dw_b = (unsigned short*)take((size_t)DI * DI * 2);
    float*          Bx   = (float*)take((size_t)M_ALL * SD * 4);
    unsigned short* Sg   = (unsigned short*)take((size_t)M_ALL * SD * 2);

    k_cvt<<<(SD * SD / 4 + 255) / 256, 256, 0, stream>>>(A,  A_b,  SD * SD / 4);
    k_cvt<<<(SD * DI / 4 + 255) / 256, 256, 0, stream>>>(Bw, Bw_b, SD * DI / 4);
    k_cvt<<<(DI * SD / 4 + 255) / 256, 256, 0, stream>>>(Cw, Cw_b, DI * SD / 4);
    k_cvt<<<(DI * DI / 4 + 255) / 256, 256, 0, stream>>>(Dw, Dw_b, DI * DI / 4);

    k_bx  <<<M_ALL / 16, 256, 0, stream>>>(x, Bw_b, Bx);
    k_scan<<<BB / 16,    256, 0, stream>>>(A_b, Bx, Sg);
    k_out <<<M_ALL / 16, 512, 0, stream>>>(x, Sg, Cw_b, Dw_b, gw, gb,
                                           gamma, beta, out);
}
